// GATConv_Adj_65128884076669
// MI455X (gfx1250) — matrix operand order repaired and checked by reading, not compiled, not run on hardware
//
#include <hip/hip_runtime.h>
#include <hip/hip_bf16.h>
#include <cmath>

#define IN_FEATS  128
#define OUT_FEATS 32

typedef __attribute__((ext_vector_type(16))) __bf16 v16bf;
typedef __attribute__((ext_vector_type(8)))  float  v8f;

// -------------------------------------------------------------------------
// Float atomic-max via monotonic integer mapping (m[] initialized to -inf).
// -------------------------------------------------------------------------
__device__ __forceinline__ void atomicMaxFloat(float* addr, float val) {
    unsigned int uv = __float_as_uint(val);
    if ((int)uv >= 0) {
        atomicMax((int*)addr, (int)uv);            // positive floats: signed int order
    } else {
        atomicMin((unsigned int*)addr, uv);        // negative floats: reversed unsigned order
    }
}

// -------------------------------------------------------------------------
// Init: out = 0, m = -inf, s = 0
// -------------------------------------------------------------------------
__global__ void gat_init_kernel(float* __restrict__ out, float* __restrict__ m,
                                float* __restrict__ s, int N) {
    int i = blockIdx.x * blockDim.x + threadIdx.x;
    if (i < N * OUT_FEATS) out[i] = 0.0f;
    if (i < N) { m[i] = -INFINITY; s[i] = 0.0f; }
}

// -------------------------------------------------------------------------
// Fused projection + attention logits.
// One wave computes a 16x32 tile of h = X @ W via two v_wmma_f32_16x16x32_bf16
// accumulators over K=128 (4 steps of K=32), then reduces el/er in-register.
//
// W is staged once per block into LDS *transposed* (Wt[c][k], bf16, 32x128 =
// 8KB) so each lane's 16-element B operand run (W[k0+16h .. k0+16h+15][col])
// is 32 contiguous, 32B-aligned bytes -> two ds_load_b128 per operand instead
// of sixteen ds_load_u16 + v_mov_b16 repacking.
// -------------------------------------------------------------------------
__global__ __launch_bounds__(256)
void gat_gemm_el_er_kernel(const float* __restrict__ feat,
                           const float* __restrict__ W,
                           const float* __restrict__ attn_l,
                           const float* __restrict__ attn_r,
                           float* __restrict__ h,
                           float* __restrict__ el,
                           float* __restrict__ er,
                           int N) {
    __shared__ __align__(32) __bf16 Wt[OUT_FEATS * IN_FEATS];  // Wt[c*128 + k] = W[k][c]
    for (int i = threadIdx.x; i < IN_FEATS * OUT_FEATS; i += blockDim.x) {
        const int k = i >> 5;      // i / 32
        const int c = i & 31;      // i % 32
        Wt[c * IN_FEATS + k] = (__bf16)W[i];
    }
    __syncthreads();

    const int wave = threadIdx.x >> 5;
    const int lane = threadIdx.x & 31;
    const int half = lane >> 4;       // K-half selector per ISA A/B layout
    const int col  = lane & 15;       // matrix row (A) / column (B,C,D) for this lane
    const int row0 = (blockIdx.x * 8 + wave) * 16;
    if (row0 >= N) return;            // wave-uniform: EXEC stays all-ones for WMMA

    const float* fr = feat + (size_t)(row0 + col) * IN_FEATS;  // A row = lane&15

    v8f acc0 = {};   // output columns 0..15
    v8f acc1 = {};   // output columns 16..31

    #pragma unroll
    for (int k0 = 0; k0 < IN_FEATS; k0 += 32) {
        // A (16x32 bf16): lane holds row col; elem i -> K = k0 + 16*half + i
        //                 (matches B's contiguous order)
        v16bf a;
        #pragma unroll
        for (int i = 0; i < 8; ++i) {
            a[i]     = (__bf16)fr[k0 + half * 16 + i];
            a[8 + i] = (__bf16)fr[k0 + half * 16 + 8 + i];
        }
        // B (32x16 bf16): lane holds column col; elem i -> K = k0 + 16*half + i.
        // Contiguous in the transposed LDS image: one 32-byte vector per operand.
        const v16bf b0 = *(const v16bf*)(Wt + (col      * IN_FEATS + k0 + half * 16));
        const v16bf b1 = *(const v16bf*)(Wt + ((col+16) * IN_FEATS + k0 + half * 16));

        acc0 = __builtin_amdgcn_wmma_f32_16x16x32_bf16(false, a, false, b0,
                                                       (short)0, acc0, false, false);
        acc1 = __builtin_amdgcn_wmma_f32_16x16x32_bf16(false, a, false, b1,
                                                       (short)0, acc1, false, false);
    }

    // C/D layout: lane (half,col) holds rows M = 8*half + r, column col (+16 for acc1)
    const float al0 = attn_l[col], al1 = attn_l[col + 16];
    const float ar0 = attn_r[col], ar1 = attn_r[col + 16];
    float pl[8], pr[8];
    #pragma unroll
    for (int r = 0; r < 8; ++r) {
        const int gr = row0 + half * 8 + r;
        h[(size_t)gr * OUT_FEATS + col]      = acc0[r];
        h[(size_t)gr * OUT_FEATS + col + 16] = acc1[r];
        pl[r] = acc0[r] * al0 + acc1[r] * al1;
        pr[r] = acc0[r] * ar0 + acc1[r] * ar1;
    }
    // reduce across the 16 lanes sharing `half` (xor masks 1,2,4,8 stay in-half)
    #pragma unroll
    for (int off = 1; off < 16; off <<= 1) {
        #pragma unroll
        for (int r = 0; r < 8; ++r) {
            pl[r] += __shfl_xor(pl[r], off, 32);
            pr[r] += __shfl_xor(pr[r], off, 32);
        }
    }
    if (col == 0) {
        #pragma unroll
        for (int r = 0; r < 8; ++r) {
            const int gr = row0 + half * 8 + r;
            el[gr] = pl[r];
            er[gr] = pr[r];
        }
    }
}

// -------------------------------------------------------------------------
// Edge pass 1: e = leaky_relu(el[u] + er[v]); segment-max into m[v]; stash e.
// -------------------------------------------------------------------------
__global__ void gat_edge_max_kernel(const int* __restrict__ src, const int* __restrict__ dst,
                                    const float* __restrict__ el, const float* __restrict__ er,
                                    float* __restrict__ m, float* __restrict__ ebuf, int E) {
    int i = blockIdx.x * blockDim.x + threadIdx.x;
    if (i >= E) return;
    const int u = src[i], v = dst[i];
    float x = el[u] + er[v];
    x = x > 0.0f ? x : 0.2f * x;
    ebuf[i] = x;
    atomicMaxFloat(&m[v], x);
}

// -------------------------------------------------------------------------
// Edge pass 2: ex = exp(e - m[v]); segment-sum into s[v]; stash ex in place.
// -------------------------------------------------------------------------
__global__ void gat_edge_exp_kernel(const int* __restrict__ dst, const float* __restrict__ m,
                                    float* __restrict__ ebuf, float* __restrict__ s, int E) {
    int i = blockIdx.x * blockDim.x + threadIdx.x;
    if (i >= E) return;
    const int v = dst[i];
    const float ex = __expf(ebuf[i] - m[v]);
    ebuf[i] = ex;
    atomicAdd(&s[v], ex);
}

// -------------------------------------------------------------------------
// Edge pass 3 (SpMM scatter): one wave per edge, one lane per output feature.
// out[v][f] += h[u][f] * (ex / s[v]); atomics resolve in L2 (h fits easily).
// -------------------------------------------------------------------------
__global__ void gat_scatter_kernel(const int* __restrict__ src, const int* __restrict__ dst,
                                   const float* __restrict__ h, const float* __restrict__ ebuf,
                                   const float* __restrict__ s, float* __restrict__ out, int E) {
    const int g    = blockIdx.x * blockDim.x + threadIdx.x;
    const int e    = g >> 5;
    const int lane = g & 31;
    if (e >= E) return;
    const int u = src[e], v = dst[e];
    const float alpha = ebuf[e] / s[v];
    atomicAdd(&out[(size_t)v * OUT_FEATS + lane], h[(size_t)u * OUT_FEATS + lane] * alpha);
}

// -------------------------------------------------------------------------
extern "C" void kernel_launch(void* const* d_in, const int* in_sizes, int n_in,
                              void* d_out, int out_size, void* d_ws, size_t ws_size,
                              hipStream_t stream) {
    const float* features = (const float*)d_in[0];
    const int*   src      = (const int*)d_in[1];
    const int*   dst      = (const int*)d_in[2];
    const float* fc_w     = (const float*)d_in[3];
    const float* attn_l   = (const float*)d_in[4];
    const float* attn_r   = (const float*)d_in[5];
    float*       out      = (float*)d_out;

    const int N = in_sizes[0] / IN_FEATS;
    const int E = in_sizes[1];

    // workspace layout (floats): h[N*32] | el[N] | er[N] | m[N] | s[N] | ebuf[E]
    float* h    = (float*)d_ws;
    float* el   = h  + (size_t)N * OUT_FEATS;
    float* er   = el + N;
    float* m    = er + N;
    float* s    = m  + N;
    float* ebuf = s  + N;

    const int TB = 256;

    // init out / m / s
    {
        int total = N * OUT_FEATS;
        gat_init_kernel<<<(total + TB - 1) / TB, TB, 0, stream>>>(out, m, s, N);
    }
    // projection + logits (8 waves/block, 16 rows per wave -> 128 rows/block)
    {
        int nblk = (N + 127) / 128;
        gat_gemm_el_er_kernel<<<nblk, TB, 0, stream>>>(features, fc_w, attn_l, attn_r,
                                                       h, el, er, N);
    }
    // edge softmax passes
    {
        int nblk = (E + TB - 1) / TB;
        gat_edge_max_kernel<<<nblk, TB, 0, stream>>>(src, dst, el, er, m, ebuf, E);
        gat_edge_exp_kernel<<<nblk, TB, 0, stream>>>(dst, m, ebuf, s, E);
    }
    // weighted scatter: 32 lanes per edge
    {
        long long total = (long long)E * 32;
        int nblk = (int)((total + TB - 1) / TB);
        gat_scatter_kernel<<<nblk, TB, 0, stream>>>(src, dst, h, ebuf, s, out, E);
    }
}
